// PointDGCN_75376676044988
// MI455X (gfx1250) — matrix operand order repaired and checked by reading, not compiled, not run on hardware
//
#include <hip/hip_runtime.h>
#include <hip/hip_bf16.h>
#include <math.h>

// ---------------------------------------------------------------------------
// PointDGCN pipeline for MI455X (gfx1250, wave32, WMMA)
// ---------------------------------------------------------------------------

#define EPS_IN 1e-5f

typedef __attribute__((ext_vector_type(16))) __bf16 v16bf;
typedef __attribute__((ext_vector_type(8)))  float  v8f;

__device__ __forceinline__ float leaky_f(float v) { return v > 0.f ? v : 0.01f * v; }
__device__ __forceinline__ bool finitef(float f) { return fabsf(f) <= 3.4e38f; }

// order-preserving float -> uint encoding for atomic max (0 == "empty" sentinel)
__device__ __forceinline__ unsigned fenc(float f) {
    unsigned u = __float_as_uint(f);
    return (u & 0x80000000u) ? ~u : (u | 0x80000000u);
}
__device__ __forceinline__ float fdec(unsigned u) {
    unsigned b = (u & 0x80000000u) ? (u & 0x7FFFFFFFu) : ~u;
    return __uint_as_float(b);
}

// ---------------------------------------------------------------------------
// Generic GEMM: C[M,Cout] = A[M,Cin](lda) @ W[Cin,Cout] + bias, bf16 WMMA
// one wave per 16x16 output tile; CIN is a compile-time constant so the
// K loop fully unrolls (native v_cvt f32->bf16 feeding v_wmma)
// ---------------------------------------------------------------------------
template <int CIN>
__global__ void gemm_bf16_wmma(const float* __restrict__ A, int lda,
                               const float* __restrict__ W, int Cout,
                               const float* __restrict__ bias,
                               float* __restrict__ C, int ldc,
                               int M, int Ntiles)
{
    int wave = (int)((blockIdx.x * (unsigned)blockDim.x + threadIdx.x) >> 5);
    int Mtiles = (M + 15) >> 4;
    if (wave >= Mtiles * Ntiles) return;

    int tm = wave / Ntiles;
    int tn = wave - tm * Ntiles;
    int lane = threadIdx.x & 31;
    int hi   = lane >> 4;          // 0: lanes 0-15, 1: lanes 16-31
    int l15  = lane & 15;
    int m0   = tm << 4;
    int n0   = tn << 4;

    int arow_i = m0 + l15; if (arow_i >= M) arow_i = M - 1;
    const float* arow = A + (size_t)arow_i * lda;
    const int kAoff = hi << 3;            // A: +0 / +8
    const int kBln  = l15;                // B: lane = N column within tile

    v8f acc = {0.f, 0.f, 0.f, 0.f, 0.f, 0.f, 0.f, 0.f};

    #pragma unroll
    for (int k0 = 0; k0 < CIN; k0 += 32) {
        // A fragment: lane holds row m0+l15; elems 0-7 = K kA..kA+7, 8-15 = K kA+16..kA+23
        const float4 a0 = *(const float4*)(arow + k0 + kAoff + 0);
        const float4 a1 = *(const float4*)(arow + k0 + kAoff + 4);
        const float4 a2 = *(const float4*)(arow + k0 + kAoff + 16);
        const float4 a3 = *(const float4*)(arow + k0 + kAoff + 20);
        // B fragment: lane holds col n0+kBln; elems 0-7 = K kA..kA+7, 8-15 = K kA+16..kA+23
        const float* wrow = W + (size_t)(k0 + kAoff) * Cout + n0 + kBln;
        const float4 b0 = make_float4(wrow[0 * Cout], wrow[1 * Cout], wrow[2 * Cout], wrow[3 * Cout]);
        const float4 b1 = make_float4(wrow[4 * Cout], wrow[5 * Cout], wrow[6 * Cout], wrow[7 * Cout]);
        const float4 b2 = make_float4(wrow[16 * Cout], wrow[17 * Cout], wrow[18 * Cout], wrow[19 * Cout]);
        const float4 b3 = make_float4(wrow[20 * Cout], wrow[21 * Cout], wrow[22 * Cout], wrow[23 * Cout]);

        if (k0 + 32 < CIN) __builtin_prefetch(arow + k0 + 32, 0, 1);

        v16bf av, bv;
        av[0]  = (__bf16)a0.x; av[1]  = (__bf16)a0.y; av[2]  = (__bf16)a0.z; av[3]  = (__bf16)a0.w;
        av[4]  = (__bf16)a1.x; av[5]  = (__bf16)a1.y; av[6]  = (__bf16)a1.z; av[7]  = (__bf16)a1.w;
        av[8]  = (__bf16)a2.x; av[9]  = (__bf16)a2.y; av[10] = (__bf16)a2.z; av[11] = (__bf16)a2.w;
        av[12] = (__bf16)a3.x; av[13] = (__bf16)a3.y; av[14] = (__bf16)a3.z; av[15] = (__bf16)a3.w;
        bv[0]  = (__bf16)b0.x; bv[1]  = (__bf16)b0.y; bv[2]  = (__bf16)b0.z; bv[3]  = (__bf16)b0.w;
        bv[4]  = (__bf16)b1.x; bv[5]  = (__bf16)b1.y; bv[6]  = (__bf16)b1.z; bv[7]  = (__bf16)b1.w;
        bv[8]  = (__bf16)b2.x; bv[9]  = (__bf16)b2.y; bv[10] = (__bf16)b2.z; bv[11] = (__bf16)b2.w;
        bv[12] = (__bf16)b3.x; bv[13] = (__bf16)b3.y; bv[14] = (__bf16)b3.z; bv[15] = (__bf16)b3.w;

        acc = __builtin_amdgcn_wmma_f32_16x16x32_bf16(
            false, av, false, bv, (short)0, acc, false, false);
    }

    const float bs = bias[n0 + l15];
    float* cc = C + (size_t)(n0 + l15);
    #pragma unroll
    for (int r = 0; r < 8; ++r) {
        int m = m0 + r + (hi << 3);        // D: VGPR r -> M=r (lo) / r+8 (hi)
        if (m < M) cc[(size_t)m * ldc] = acc[r] + bs;
    }
}

// ---------------------------------------------------------------------------
// Small elementwise / reduction kernels
// ---------------------------------------------------------------------------
__global__ void cnt_kernel(const int* __restrict__ batch, int N, float* __restrict__ cnt) {
    int n = blockIdx.x * blockDim.x + threadIdx.x;
    if (n < N) atomicAdd(&cnt[batch[n]], 1.f);
}

__global__ void lin0_kernel(const float* __restrict__ x, const float* __restrict__ W0,
                            const float* __restrict__ b0, float* __restrict__ out, int N) {
    int i = blockIdx.x * blockDim.x + threadIdx.x;
    if (i >= N * 64) return;
    int n = i >> 6, c = i & 63;
    float s = b0[c];
    const float* xr = x + (size_t)n * 15;
    #pragma unroll
    for (int h = 0; h < 15; ++h) s += xr[h] * W0[h * 64 + c];
    out[i] = s;
}

__global__ void stats_kernel(const float* __restrict__ v, const int* __restrict__ batch,
                             int N, int C, float* __restrict__ sum, float* __restrict__ sumsq) {
    int i = blockIdx.x * blockDim.x + threadIdx.x;
    if (i >= N * C) return;
    int n = i / C, c = i - n * C;
    float f = v[i];
    int b = batch[n];
    atomicAdd(&sum[b * C + c], f);
    atomicAdd(&sumsq[b * C + c], f * f);
}

__global__ void meanvar_kernel(const float* __restrict__ sum, const float* __restrict__ sumsq,
                               const float* __restrict__ cnt, int C,
                               float* __restrict__ mean, float* __restrict__ rstd) {
    int i = blockIdx.x * blockDim.x + threadIdx.x;
    if (i >= 8 * C) return;
    int b = i / C;
    float ct = fmaxf(cnt[b], 1.f);
    float m = sum[i] / ct;
    float var = sumsq[i] / ct - m * m;
    mean[i] = m;
    rstd[i] = rsqrtf(var + EPS_IN);
}

__global__ void norm_leaky_kernel(const float* __restrict__ v, const int* __restrict__ batch,
                                  int N, int C, const float* __restrict__ mean,
                                  const float* __restrict__ rstd,
                                  float* __restrict__ out, int ldo, int col0) {
    int i = blockIdx.x * blockDim.x + threadIdx.x;
    if (i >= N * C) return;
    int n = i / C, c = i - n * C;
    int b = batch[n];
    float f = (v[i] - mean[b * C + c]) * rstd[b * C + c];
    out[(size_t)n * ldo + col0 + c] = leaky_f(f);
}

__global__ void segmax_kernel(const float* __restrict__ v, const int* __restrict__ batch,
                              int N, unsigned* __restrict__ genc) {
    int i = blockIdx.x * blockDim.x + threadIdx.x;
    if (i >= N * 256) return;
    int n = i >> 8, c = i & 255;
    atomicMax(&genc[batch[n] * 256 + c], fenc(v[i]));
}

// B=8 tiny MLP: genc -> g -> g1(256) -> g2(128) -> T[8,64,64] (+I)
__global__ void mlp_kernel(const unsigned* __restrict__ genc,
                           const float* __restrict__ fw1, const float* __restrict__ fb1,
                           const float* __restrict__ fw2, const float* __restrict__ fb2,
                           const float* __restrict__ fw3, const float* __restrict__ fb3,
                           float* __restrict__ T) {
    __shared__ float g[8 * 256], g1[8 * 256], g2[8 * 128];
    int t = threadIdx.x;
    for (int i = t; i < 8 * 256; i += 256) {
        unsigned u = genc[i];
        float f = (u == 0u) ? 0.f : fdec(u);
        if (!finitef(f)) f = 0.f;
        g[i] = f;
    }
    __syncthreads();
    for (int i = t; i < 8 * 256; i += 256) {
        int b = i >> 8, j = i & 255;
        float s = fb1[j];
        for (int h = 0; h < 256; ++h) s += g[b * 256 + h] * fw1[h * 256 + j];
        g1[i] = leaky_f(s);
    }
    __syncthreads();
    for (int i = t; i < 8 * 128; i += 256) {
        int b = i >> 7, j = i & 127;
        float s = fb2[j];
        for (int h = 0; h < 256; ++h) s += g1[b * 256 + h] * fw2[h * 128 + j];
        g2[i] = leaky_f(s);
    }
    __syncthreads();
    for (int i = t; i < 8 * 4096; i += 256) {
        int b = i >> 12, j = i & 4095;
        float s = fb3[j];
        for (int h = 0; h < 128; ++h) s += g2[b * 128 + h] * fw3[h * 4096 + j];
        if ((j >> 6) == (j & 63)) s += 1.f;   // + identity
        T[i] = s;
    }
}

__global__ void xt_kernel(const float* __restrict__ x0, const float* __restrict__ T,
                          const int* __restrict__ batch, int N, float* __restrict__ out) {
    int i = blockIdx.x * blockDim.x + threadIdx.x;
    if (i >= N * 64) return;
    int n = i >> 6, c = i & 63;
    const float* Tb = T + (size_t)batch[n] * 4096 + c;
    const float* xr = x0 + (size_t)n * 64;
    float s = 0.f;
    #pragma unroll 8
    for (int h = 0; h < 64; ++h) s += xr[h] * Tb[h * 64];
    out[i] = s;
}

// Wcat[64,128] = [W_top - W_bot | W_bot], bcat = [b | 0]
__global__ void wcat_kernel(const float* __restrict__ W, const float* __restrict__ b,
                            float* __restrict__ Wcat, float* __restrict__ bcat) {
    int i = blockIdx.x * blockDim.x + threadIdx.x;
    if (i >= 64 * 128) return;
    int k = i >> 7, n = i & 127;
    if (n < 64) Wcat[i] = W[k * 64 + n] - W[(64 + k) * 64 + n];
    else        Wcat[i] = W[(64 + k) * 64 + (n - 64)];
    if (i < 128) bcat[i] = (i < 64) ? b[i] : 0.f;
}

// msg_e = A[dst] + Bn[src]; atomic max-aggregate at dst (one wave per edge)
__global__ void edge_kernel(const int* __restrict__ ei, int E,
                            const float* __restrict__ AB, unsigned* __restrict__ maxu) {
    int gid = blockIdx.x * blockDim.x + threadIdx.x;
    int e = gid >> 5;
    if (e >= E) return;
    int lane = gid & 31;
    int src = ei[e];
    int dst = ei[E + e];
    float2 a  = ((const float2*)(AB + (size_t)dst * 128))[lane];
    float2 bb = ((const float2*)(AB + (size_t)src * 128 + 64))[lane];
    unsigned* mrow = maxu + (size_t)dst * 64 + (lane << 1);
    atomicMax(&mrow[0], fenc(a.x + bb.x));
    atomicMax(&mrow[1], fenc(a.y + bb.y));
}

__global__ void decode_stats_kernel(const unsigned* __restrict__ maxu,
                                    const int* __restrict__ batch, int N,
                                    float* __restrict__ sum, float* __restrict__ sumsq,
                                    float* __restrict__ dec) {
    int i = blockIdx.x * blockDim.x + threadIdx.x;
    if (i >= N * 64) return;
    int n = i >> 6, c = i & 63;
    unsigned u = maxu[i];
    float f = (u == 0u) ? 0.f : fdec(u);
    if (!finitef(f)) f = 0.f;      // nodes with no in-edges -> 0
    dec[i] = f;
    int b = batch[n];
    atomicAdd(&sum[b * 64 + c], f);
    atomicAdd(&sumsq[b * 64 + c], f * f);
}

__global__ void gate_kernel(const float* __restrict__ xc, const float* __restrict__ Wg,
                            const float* __restrict__ bg, const int* __restrict__ batch,
                            int N, float* __restrict__ gate, unsigned* __restrict__ gmaxe) {
    int n = blockIdx.x * blockDim.x + threadIdx.x;
    if (n >= N) return;
    const float* r = xc + (size_t)n * 256;
    float s = bg[0];
    #pragma unroll 8
    for (int c = 0; c < 256; ++c) s += r[c] * Wg[c];
    gate[n] = s;
    atomicMax(&gmaxe[batch[n]], fenc(s));
}

__global__ void expden_kernel(float* __restrict__ gate, const int* __restrict__ batch,
                              int N, const unsigned* __restrict__ gmaxe, float* __restrict__ den) {
    int n = blockIdx.x * blockDim.x + threadIdx.x;
    if (n >= N) return;
    int b = batch[n];
    unsigned u = gmaxe[b];
    float gm = (u == 0u) ? 0.f : fdec(u);
    if (!finitef(gm)) gm = 0.f;
    float ex = expf(gate[n] - gm);
    gate[n] = ex;
    atomicAdd(&den[b], ex);
}

__global__ void pooled_kernel(const float* __restrict__ xc, const float* __restrict__ gate,
                              const int* __restrict__ batch, int N, float* __restrict__ pooled) {
    int i = blockIdx.x * blockDim.x + threadIdx.x;
    if (i >= N * 256) return;
    int n = i >> 8, c = i & 255;
    atomicAdd(&pooled[batch[n] * 256 + c], gate[n] * xc[i]);
}

__global__ void final_kernel(const float* __restrict__ xc, const float* __restrict__ pooled,
                             const float* __restrict__ den, const int* __restrict__ batch,
                             int N, float* __restrict__ out) {
    int i = blockIdx.x * blockDim.x + threadIdx.x;
    if (i >= N * 512) return;
    int n = i >> 9, c = i & 511;
    if (c < 256) out[i] = xc[(size_t)n * 256 + c];
    else {
        int b = batch[n];
        out[i] = pooled[b * 256 + (c - 256)] / (den[b] + 1e-16f);
    }
}

// ---------------------------------------------------------------------------
static inline int gdiv(long a, long b) { return (int)((a + b - 1) / b); }

extern "C" void kernel_launch(void* const* d_in, const int* in_sizes, int n_in,
                              void* d_out, int out_size, void* d_ws, size_t ws_size,
                              hipStream_t stream)
{
    (void)n_in; (void)out_size; (void)ws_size;
    const float* x     = (const float*)d_in[0];
    const int*   ei    = (const int*)d_in[1];
    const int*   batch = (const int*)d_in[2];
    const float* W0  = (const float*)d_in[3];  const float* b0  = (const float*)d_in[4];
    const float* sw1 = (const float*)d_in[5];  const float* sb1 = (const float*)d_in[6];
    const float* sw2 = (const float*)d_in[7];  const float* sb2 = (const float*)d_in[8];
    const float* sw3 = (const float*)d_in[9];  const float* sb3 = (const float*)d_in[10];
    const float* fw1 = (const float*)d_in[11]; const float* fb1 = (const float*)d_in[12];
    const float* fw2 = (const float*)d_in[13]; const float* fb2 = (const float*)d_in[14];
    const float* fw3 = (const float*)d_in[15]; const float* fb3 = (const float*)d_in[16];
    const float* Wk[4] = {(const float*)d_in[17], (const float*)d_in[19],
                          (const float*)d_in[21], (const float*)d_in[23]};
    const float* bk[4] = {(const float*)d_in[18], (const float*)d_in[20],
                          (const float*)d_in[22], (const float*)d_in[24]};
    const float* Wg = (const float*)d_in[25]; const float* bg = (const float*)d_in[26];

    const int N = in_sizes[0] / 15;
    const int E = in_sizes[1] / 2;

    // ---- workspace layout ----
    char* wsb = (char*)d_ws;
    size_t off = 0;
    auto take = [&](size_t bytes) -> char* {
        char* p = wsb + off;
        off += (bytes + 255) & ~(size_t)255;
        return p;
    };
    float*    x0    = (float*)take((size_t)N * 64 * 4);   // later reused as maxu (u32)
    float*    bufA  = (float*)take((size_t)N * 256 * 4);  // h1/h3/AB
    float*    bufB  = (float*)take((size_t)N * 128 * 4);  // h2/xt/decoded
    float*    xc    = (float*)take((size_t)N * 256 * 4);
    float*    T     = (float*)take((size_t)8 * 4096 * 4);
    float*    Wcat  = (float*)take((size_t)64 * 128 * 4);
    float*    bcat  = (float*)take((size_t)128 * 4);
    float*    sum   = (float*)take((size_t)8 * 256 * 4);
    float*    sumsq = (float*)take((size_t)8 * 256 * 4);
    float*    mean  = (float*)take((size_t)8 * 256 * 4);
    float*    rstd  = (float*)take((size_t)8 * 256 * 4);
    float*    cnt   = (float*)take((size_t)8 * 4);
    unsigned* genc  = (unsigned*)take((size_t)8 * 256 * 4);
    float*    gate  = (float*)take((size_t)N * 4);
    unsigned* gmaxe = (unsigned*)take((size_t)8 * 4);
    float*    den   = (float*)take((size_t)8 * 4);
    float*    pooled= (float*)take((size_t)8 * 256 * 4);

    auto gemm = [&](const float* A, int lda, const float* W, int Cout,
                    const float* bias, float* C, int ldc, int Cin) {
        int Mtiles = (N + 15) / 16, Ntiles = Cout / 16;
        long threads = (long)Mtiles * Ntiles * 32;
        if (Cin == 64)
            gemm_bf16_wmma<64><<<gdiv(threads, 256), 256, 0, stream>>>(
                A, lda, W, Cout, bias, C, ldc, N, Ntiles);
        else
            gemm_bf16_wmma<128><<<gdiv(threads, 256), 256, 0, stream>>>(
                A, lda, W, Cout, bias, C, ldc, N, Ntiles);
    };
    auto inorm = [&](float* v, int C, float* outp, int ldo, int col0) {
        hipMemsetAsync(sum,   0, 8 * 256 * 4, stream);
        hipMemsetAsync(sumsq, 0, 8 * 256 * 4, stream);
        stats_kernel<<<gdiv((long)N * C, 256), 256, 0, stream>>>(v, batch, N, C, sum, sumsq);
        meanvar_kernel<<<gdiv(8L * C, 256), 256, 0, stream>>>(sum, sumsq, cnt, C, mean, rstd);
        norm_leaky_kernel<<<gdiv((long)N * C, 256), 256, 0, stream>>>(
            v, batch, N, C, mean, rstd, outp, ldo, col0);
    };

    // ---- per-graph node counts ----
    hipMemsetAsync(cnt, 0, 8 * 4, stream);
    cnt_kernel<<<gdiv(N, 256), 256, 0, stream>>>(batch, N, cnt);

    // ---- x0 = leaky(inorm(x @ W0 + b0)) ----
    lin0_kernel<<<gdiv((long)N * 64, 256), 256, 0, stream>>>(x, W0, b0, x0, N);
    inorm(x0, 64, x0, 64, 0);

    // ---- STN chain (bf16 WMMA GEMMs) ----
    gemm(x0, 64, sw1, 64, sb1, bufA, 64, 64);       inorm(bufA, 64, bufA, 64, 0);
    gemm(bufA, 64, sw2, 128, sb2, bufB, 128, 64);   inorm(bufB, 128, bufB, 128, 0);
    gemm(bufB, 128, sw3, 256, sb3, bufA, 256, 128); inorm(bufA, 256, bufA, 256, 0);

    // ---- global max pool + tiny MLP -> T ----
    hipMemsetAsync(genc, 0, 8 * 256 * 4, stream);
    segmax_kernel<<<gdiv((long)N * 256, 256), 256, 0, stream>>>(bufA, batch, N, genc);
    mlp_kernel<<<1, 256, 0, stream>>>(genc, fw1, fb1, fw2, fb2, fw3, fb3, T);

    // ---- xt = x0 @ T[batch] ----
    xt_kernel<<<gdiv((long)N * 64, 256), 256, 0, stream>>>(x0, T, batch, N, bufB);

    // ---- 4 EdgeConv blocks ----
    const float* xin = bufB; int ldin = 64;
    unsigned* maxu = (unsigned*)x0;   // x0 dead after xt
    for (int k = 0; k < 4; ++k) {
        wcat_kernel<<<gdiv(64L * 128, 256), 256, 0, stream>>>(Wk[k], bk[k], Wcat, bcat);
        // AB[:,0:64] = x@(Wtop-Wbot)+b,  AB[:,64:128] = x@Wbot   (WMMA)
        gemm(xin, ldin, Wcat, 128, bcat, bufA, 128, 64);
        hipMemsetAsync(maxu, 0, (size_t)N * 64 * 4, stream);
        edge_kernel<<<gdiv((long)E * 32, 256), 256, 0, stream>>>(ei, E, bufA, maxu);
        hipMemsetAsync(sum,   0, 8 * 256 * 4, stream);
        hipMemsetAsync(sumsq, 0, 8 * 256 * 4, stream);
        decode_stats_kernel<<<gdiv((long)N * 64, 256), 256, 0, stream>>>(
            maxu, batch, N, sum, sumsq, bufB);
        meanvar_kernel<<<gdiv(8L * 64, 256), 256, 0, stream>>>(sum, sumsq, cnt, 64, mean, rstd);
        norm_leaky_kernel<<<gdiv((long)N * 64, 256), 256, 0, stream>>>(
            bufB, batch, N, 64, mean, rstd, xc, 256, k * 64);
        xin = xc + k * 64; ldin = 256;
    }

    // ---- attention pooling + output ----
    hipMemsetAsync(gmaxe, 0, 8 * 4, stream);
    gate_kernel<<<gdiv(N, 256), 256, 0, stream>>>(xc, Wg, bg, batch, N, gate, gmaxe);
    hipMemsetAsync(den, 0, 8 * 4, stream);
    expden_kernel<<<gdiv(N, 256), 256, 0, stream>>>(gate, batch, N, gmaxe, den);
    hipMemsetAsync(pooled, 0, 8 * 256 * 4, stream);
    pooled_kernel<<<gdiv((long)N * 256, 256), 256, 0, stream>>>(xc, gate, batch, N, pooled);
    final_kernel<<<gdiv((long)N * 512, 256), 256, 0, stream>>>(
        xc, pooled, den, batch, N, (float*)d_out);
}
